// GATLayer_49727131353449
// MI455X (gfx1250) — matrix operand order repaired and checked by reading, not compiled, not run on hardware
//
#include <hip/hip_runtime.h>
#include <hip/hip_bf16.h>

#define NN 8192
#define IN_DIM 256
#define NUM_HEADS 4
#define HEAD_DIM 64
#define NEG_SLOPE 0.2f
#define LOG2E 1.4426950408889634f

typedef __attribute__((ext_vector_type(16))) _Float16 v16h;
typedef __attribute__((ext_vector_type(2)))  __fp16   v2hf;   // cvt_pkrtz result type
typedef __attribute__((ext_vector_type(8)))  float    v8f;

// ---------------------------------------------------------------------------
// Kernel 1: Wh = h @ W per head; store WhT[h][d][n] as f16 (B-operand layout:
// column-major over d so each WMMA lane loads 16 contiguous K=j values),
// plus s_i' = (Wh.a1)*log2e, s_j' = (Wh.a2)*log2e via LDS reduction.
// Everything downstream works in the log2 domain so the inner loop's exp is
// a bare v_exp_f32.
// One block per node n; 256 threads = (head = tid/64, d = tid%64).
// ---------------------------------------------------------------------------
__global__ __launch_bounds__(256) void wh_kernel(
    const float* __restrict__ h, const float* __restrict__ W,
    const float* __restrict__ a, _Float16* __restrict__ WhT,
    float* __restrict__ s_i, float* __restrict__ s_j)
{
    const int n   = blockIdx.x;
    const int tid = threadIdx.x;
    const int hd  = tid >> 6;     // head
    const int d   = tid & 63;     // output feature

    __shared__ float sh[IN_DIM];
    __shared__ float r1[256];
    __shared__ float r2[256];

    sh[tid] = h[(size_t)n * IN_DIM + tid];
    __syncthreads();

    const float* wp = W + ((size_t)hd * IN_DIM) * HEAD_DIM + d;
    float acc = 0.f;
#pragma unroll 8
    for (int i = 0; i < IN_DIM; ++i)
        acc = __builtin_fmaf(sh[i], wp[(size_t)i * HEAD_DIM], acc);

    WhT[((size_t)(hd * HEAD_DIM + d)) * NN + n] = (_Float16)acc;

    r1[tid] = acc * a[d];
    r2[tid] = acc * a[HEAD_DIM + d];
    __syncthreads();
#pragma unroll
    for (int s = 32; s > 0; s >>= 1) {
        if (d < s) { r1[tid] += r1[tid + s]; r2[tid] += r2[tid + s]; }
        __syncthreads();
    }
    if (d == 0) {
        s_i[hd * NN + n] = r1[tid] * LOG2E;   // log2-domain
        s_j[hd * NN + n] = r2[tid] * LOG2E;   // log2-domain
    }
}

// ---------------------------------------------------------------------------
// Kernel 2: per-head mj' = max_n s_j'[h,n]; mrow[h,i] = -lrelu'(s_i'+mj').
// (lrelu in the scaled domain: max(x, 0.2x), valid since log2e > 0.)
// lrelu is monotone, so this upper-bounds the masked row max; softmax is
// shift-invariant => exact result, no online rescaling needed.
// ---------------------------------------------------------------------------
__global__ __launch_bounds__(256) void mrow_kernel(
    const float* __restrict__ s_i, const float* __restrict__ s_j,
    float* __restrict__ mrow)
{
    const int hd  = blockIdx.x;
    const int tid = threadIdx.x;
    __shared__ float red[256];

    float mx = -1e30f;
    for (int n = tid; n < NN; n += 256) mx = fmaxf(mx, s_j[hd * NN + n]);
    red[tid] = mx;
    __syncthreads();
#pragma unroll
    for (int s = 128; s > 0; s >>= 1) {
        if (tid < s) red[tid] = fmaxf(red[tid], red[tid + s]);
        __syncthreads();
    }
    const float mj = red[0];
    for (int n = tid; n < NN; n += 256) {
        float x = s_i[hd * NN + n] + mj;
        x = fmaxf(x, NEG_SLOPE * x);          // lrelu (scaled domain)
        mrow[hd * NN + n] = -x;               // pre-negated
    }
}

// ---------------------------------------------------------------------------
// Kernel 3: masked softmax + att@Wh, single pass over adj (256 MB -> the HBM
// roofline), contraction via v_wmma_f32_16x16x32_f16.
// Grid: NN/32 blocks; block: 256 threads = 8 waves.
// Wave w -> head = w&3, row-tile = w>>2 (16 rows each). Each wave owns the
// full 16x64 output tile of its (rows, head): 4 WMMA accumulators, plus a
// 5th WMMA against an all-ones B that accumulates the softmax denominator
// directly in C layout (c4[r] = rowsum of row r+8*half for every lane).
//
// Per element (log2 domain):
//   arg = max(sj' + c1, fma(sj', 0.2, c2)),  c1 = si'+nm2, c2 = 0.2*si'+nm2
//   p   = adj ? exp2(arg) : 0           -> one v_exp_f32, one cndmask
//   pack pairs with v_cvt_pk_rtz_f16_f32
//
// A-operand (P tile, 16x32 f16) lane layout (contiguous K, same order as B):
//   lane L: row M = L%16, half = L/16;
//   elements 0..7  = K = 16*half + 0..7
//   elements 8..15 = K = 16*half + 8..15
// B-operand (WhT tile, 32x16): lane L holds column N = L%16,
//   elements 0..15 = K = 16*half + 0..15 (contiguous -> one 32B load).
// ---------------------------------------------------------------------------
__global__ __launch_bounds__(256) void gat_attn_kernel(
    const int* __restrict__ adj, const _Float16* __restrict__ WhT,
    const float* __restrict__ s_i, const float* __restrict__ s_j,
    const float* __restrict__ mrow, float* __restrict__ out)
{
    const int tid  = threadIdx.x;
    const int w    = tid >> 5;
    const int lane = tid & 31;
    const int head = w & 3;
    const int rt   = w >> 2;
    const int M    = lane & 15;
    const int half = lane >> 4;

    const int i0  = blockIdx.x * 32 + rt * 16;
    const int row = i0 + M;

    const float si  = s_i[head * NN + row];    // log2 domain
    const float nm2 = mrow[head * NN + row];   // -lrelu'(si+mj)
    const float cA  = si + nm2;                // per-lane consts
    const float cB  = __builtin_fmaf(si, NEG_SLOPE, nm2);

    const float* __restrict__ sjh    = s_j + head * NN;
    const int*   __restrict__ adjrow = adj + (size_t)row * NN;
    const _Float16* __restrict__ bhead = WhT + (size_t)(head * HEAD_DIM) * NN;

    v16h bones;
#pragma unroll
    for (int e = 0; e < 16; ++e) bones[e] = (_Float16)1.0f;

    v8f c0 = {}, c1 = {}, c2 = {}, c3 = {}, c4 = {};

#pragma unroll 2
    for (int j0 = 0; j0 < NN; j0 += 32) {
        // B tiles first: in flight during the long A-computation VALU chain.
        const int bj = j0 + 16 * half;
        const v16h b0 = *(const v16h*)(bhead + (size_t)(0 * 16 + M) * NN + bj);
        const v16h b1 = *(const v16h*)(bhead + (size_t)(1 * 16 + M) * NN + bj);
        const v16h b2 = *(const v16h*)(bhead + (size_t)(2 * 16 + M) * NN + bj);
        const v16h b3 = *(const v16h*)(bhead + (size_t)(3 * 16 + M) * NN + bj);

        const int jA = j0 + 16 * half;
        const int jB = j0 + 16 * half + 8;

        const int4 ad0 = *(const int4*)(adjrow + jA);
        const int4 ad1 = *(const int4*)(adjrow + jA + 4);
        const int4 ad2 = *(const int4*)(adjrow + jB);
        const int4 ad3 = *(const int4*)(adjrow + jB + 4);
        const float4 sj0 = *(const float4*)(sjh + jA);
        const float4 sj1 = *(const float4*)(sjh + jA + 4);
        const float4 sj2 = *(const float4*)(sjh + jB);
        const float4 sj3 = *(const float4*)(sjh + jB + 4);

        auto pv = [&](int ad, float sj) -> float {
            float u   = sj + cA;                              // dual-issue add
            float v   = __builtin_fmaf(sj, NEG_SLOPE, cB);    // dual-issue fma
            float arg = fmaxf(u, v);                          // lrelu - m
            float p   = __builtin_amdgcn_exp2f(arg);          // v_exp_f32
            return ad ? p : 0.f;                              // cndmask
        };

        union { v16h v; v2hf h[8]; } A;
        A.h[0] = __builtin_amdgcn_cvt_pkrtz(pv(ad0.x, sj0.x), pv(ad0.y, sj0.y));
        A.h[1] = __builtin_amdgcn_cvt_pkrtz(pv(ad0.z, sj0.z), pv(ad0.w, sj0.w));
        A.h[2] = __builtin_amdgcn_cvt_pkrtz(pv(ad1.x, sj1.x), pv(ad1.y, sj1.y));
        A.h[3] = __builtin_amdgcn_cvt_pkrtz(pv(ad1.z, sj1.z), pv(ad1.w, sj1.w));
        A.h[4] = __builtin_amdgcn_cvt_pkrtz(pv(ad2.x, sj2.x), pv(ad2.y, sj2.y));
        A.h[5] = __builtin_amdgcn_cvt_pkrtz(pv(ad2.z, sj2.z), pv(ad2.w, sj2.w));
        A.h[6] = __builtin_amdgcn_cvt_pkrtz(pv(ad3.x, sj3.x), pv(ad3.y, sj3.y));
        A.h[7] = __builtin_amdgcn_cvt_pkrtz(pv(ad3.z, sj3.z), pv(ad3.w, sj3.w));

        c0 = __builtin_amdgcn_wmma_f32_16x16x32_f16(false, A.v, false, b0, (short)0, c0, false, false);
        c1 = __builtin_amdgcn_wmma_f32_16x16x32_f16(false, A.v, false, b1, (short)0, c1, false, false);
        c2 = __builtin_amdgcn_wmma_f32_16x16x32_f16(false, A.v, false, b2, (short)0, c2, false, false);
        c3 = __builtin_amdgcn_wmma_f32_16x16x32_f16(false, A.v, false, b3, (short)0, c3, false, false);
        c4 = __builtin_amdgcn_wmma_f32_16x16x32_f16(false, A.v, false, bones, (short)0, c4, false, false);
    }

    // C layout: VGPR r, lane L -> row = r + 8*(L/16), col = L%16.
    // c4[r] already holds that row's softmax denominator in every lane.
#pragma unroll
    for (int r = 0; r < 8; ++r) {
        const float inv = __builtin_amdgcn_rcpf(c4[r]);
        const int orow  = i0 + r + 8 * half;
        float* op = out + (size_t)orow * (NUM_HEADS * HEAD_DIM) + head * HEAD_DIM + M;
        op[0]  = c0[r] * inv;
        op[16] = c1[r] * inv;
        op[32] = c2[r] * inv;
        op[48] = c3[r] * inv;
    }
}

extern "C" void kernel_launch(void* const* d_in, const int* in_sizes, int n_in,
                              void* d_out, int out_size, void* d_ws, size_t ws_size,
                              hipStream_t stream) {
    const float* h   = (const float*)d_in[0];
    const int*   adj = (const int*)d_in[1];
    const float* W   = (const float*)d_in[2];
    const float* a   = (const float*)d_in[3];
    float* out = (float*)d_out;

    char* ws = (char*)d_ws;
    _Float16* WhT = (_Float16*)ws;                                   // 4 MB
    float* s_i    = (float*)(ws + (4u << 20));                       // 128 KB
    float* s_j    = (float*)(ws + (4u << 20) + (128u << 10));        // 128 KB
    float* mrow   = (float*)(ws + (4u << 20) + (256u << 10));        // 128 KB

    wh_kernel<<<NN, 256, 0, stream>>>(h, W, a, WhT, s_i, s_j);
    mrow_kernel<<<NUM_HEADS, 256, 0, stream>>>(s_i, s_j, mrow);
    gat_attn_kernel<<<NN / 32, 256, 0, stream>>>(adj, WhT, s_i, s_j, mrow, out);
}
